// MostSimilarImage_19009525252280
// MI455X (gfx1250) — matrix operand order repaired and checked by reading, not compiled, not run on hardware
//
#include <hip/hip_runtime.h>
#include <hip/hip_bf16.h>

#define QB    64
#define QD    512
#define VOCAB 5000
#define LTGT  100
#define LDB   120

typedef __bf16 v16bf __attribute__((ext_vector_type(16)));
typedef __bf16 v2bf  __attribute__((ext_vector_type(2)));
typedef float  v8f   __attribute__((ext_vector_type(8)));
typedef float  v4f   __attribute__((ext_vector_type(4)));

union AFrag { uint4 u[2]; v16bf v; };

__device__ __forceinline__ unsigned short f2bf_bits(float f) {
    unsigned u = __builtin_bit_cast(unsigned, f);
    unsigned r = u + 0x7FFFu + ((u >> 16) & 1u);   // round-to-nearest-even
    return (unsigned short)(r >> 16);
}

// Convert 16 f32 (4 x v4f) -> v16bf, using the native packed cvt when present.
__device__ __forceinline__ v16bf cvt16(const v4f f0, const v4f f1,
                                       const v4f f2, const v4f f3) {
    v16bf b;
#if __has_builtin(__builtin_amdgcn_cvt_pk_bf16_f32)
    v2bf p;
    p = __builtin_amdgcn_cvt_pk_bf16_f32(f0[0], f0[1]); b[0]  = p[0]; b[1]  = p[1];
    p = __builtin_amdgcn_cvt_pk_bf16_f32(f0[2], f0[3]); b[2]  = p[0]; b[3]  = p[1];
    p = __builtin_amdgcn_cvt_pk_bf16_f32(f1[0], f1[1]); b[4]  = p[0]; b[5]  = p[1];
    p = __builtin_amdgcn_cvt_pk_bf16_f32(f1[2], f1[3]); b[6]  = p[0]; b[7]  = p[1];
    p = __builtin_amdgcn_cvt_pk_bf16_f32(f2[0], f2[1]); b[8]  = p[0]; b[9]  = p[1];
    p = __builtin_amdgcn_cvt_pk_bf16_f32(f2[2], f2[3]); b[10] = p[0]; b[11] = p[1];
    p = __builtin_amdgcn_cvt_pk_bf16_f32(f3[0], f3[1]); b[12] = p[0]; b[13] = p[1];
    p = __builtin_amdgcn_cvt_pk_bf16_f32(f3[2], f3[3]); b[14] = p[0]; b[15] = p[1];
#else
    b[0]=(__bf16)f0[0];  b[1]=(__bf16)f0[1];  b[2]=(__bf16)f0[2];  b[3]=(__bf16)f0[3];
    b[4]=(__bf16)f1[0];  b[5]=(__bf16)f1[1];  b[6]=(__bf16)f1[2];  b[7]=(__bf16)f1[3];
    b[8]=(__bf16)f2[0];  b[9]=(__bf16)f2[1];  b[10]=(__bf16)f2[2]; b[11]=(__bf16)f2[3];
    b[12]=(__bf16)f3[0]; b[13]=(__bf16)f3[1]; b[14]=(__bf16)f3[2]; b[15]=(__bf16)f3[3];
#endif
    return b;
}

// ---------------------------------------------------------------------------
// Kernel 1: normalize the 64 query rows -> bf16 in workspace; init best[].
// ---------------------------------------------------------------------------
__global__ void __launch_bounds__(256)
norm_kernel(const float* __restrict__ feats,
            unsigned short* __restrict__ qbf,
            unsigned long long* __restrict__ best) {
    __shared__ float red[256];
    const int b = blockIdx.x;
    const float* row = feats + (size_t)b * QD;
    float ss = 0.0f;
    for (int i = threadIdx.x; i < QD; i += 256) {
        float v = row[i];
        ss += v * v;
    }
    red[threadIdx.x] = ss;
    __syncthreads();
    for (int s = 128; s > 0; s >>= 1) {
        if (threadIdx.x < s) red[threadIdx.x] += red[threadIdx.x + s];
        __syncthreads();
    }
    float inv = 1.0f / fmaxf(sqrtf(red[0]), 1e-8f);
    for (int i = threadIdx.x; i < QD; i += 256)
        qbf[(size_t)b * QD + i] = f2bf_bits(row[i] * inv);
    if (b == 0 && threadIdx.x < QB) best[threadIdx.x] = 0ull;
}

// ---------------------------------------------------------------------------
// Kernel 2: streamed bf16 WMMA similarity + running argmax.
//   Each wave: 32 DB rows (2 N-tiles) x all 64 queries (4 M-tiles) per pass.
//   A (queries) from padded LDS (loop-invariant address, immediate offsets),
//   B (db rows) converted f32->bf16 in registers from the HBM stream.
// ---------------------------------------------------------------------------
#define QSTRIDE 65   // row stride in 16B chunks (64 data + 1 pad) -> bank-safe

__global__ void __launch_bounds__(256)
sim_kernel(const float* __restrict__ db,
           const uint4* __restrict__ qg,
           unsigned long long* __restrict__ best,
           int nblk) {
    __shared__ uint4 qlds[QB * QSTRIDE];          // 66560 B
    for (int c = threadIdx.x; c < QB * 64; c += 256) {
        int row = c >> 6, kc = c & 63;
        qlds[row * QSTRIDE + kc] = qg[c];
    }
    __syncthreads();

    const int lane = threadIdx.x & 31;
    const int half = lane >> 4;                   // K-half of the fragment
    const int nrow = lane & 15;                   // N index / row within tile
    const int wave = (blockIdx.x << 3) | (threadIdx.x >> 5);
    const int nwav = gridDim.x << 3;

    // Loop-invariant LDS chunk base per M-tile (A row = m*16 + nrow).
    int abase[4];
#pragma unroll
    for (int m = 0; m < 4; ++m)
        abase[m] = ((m << 4) | nrow) * QSTRIDE + (half << 1);

    float bestv[4][8];
    int   besti[4][8];
#pragma unroll
    for (int m = 0; m < 4; ++m)
#pragma unroll
        for (int r = 0; r < 8; ++r) { bestv[m][r] = -3.0e38f; besti[m][r] = 0; }

    // Contiguous balanced partition of the 32-row blocks.
    const int start = (int)(((long long)wave * nblk) / nwav);
    const int stop  = (int)(((long long)(wave + 1) * nblk) / nwav);

    for (int blk = start; blk < stop; ++blk) {
        v8f acc[2][4];
#pragma unroll
        for (int t = 0; t < 2; ++t)
#pragma unroll
            for (int m = 0; m < 4; ++m)
                acc[t][m] = (v8f){0.f,0.f,0.f,0.f,0.f,0.f,0.f,0.f};
        float ssq0 = 0.0f, ssq1 = 0.0f;

        // B layout: lanes0-15 hold K=k0..k0+15, lanes16-31 hold K=k0+16..31.
        const float* dr0 = db + ((size_t)(blk * 32 + nrow)) * QD + half * 16;
        const float* dr1 = dr0 + (size_t)16 * QD;

#pragma unroll 2
        for (int k0 = 0; k0 < QD; k0 += 32) {
            const v4f* p0 = (const v4f*)(dr0 + k0);
            const v4f* p1 = (const v4f*)(dr1 + k0);
            v4f a0 = __builtin_nontemporal_load(p0 + 0);
            v4f a1 = __builtin_nontemporal_load(p0 + 1);
            v4f a2 = __builtin_nontemporal_load(p0 + 2);
            v4f a3 = __builtin_nontemporal_load(p0 + 3);
            v4f c0 = __builtin_nontemporal_load(p1 + 0);
            v4f c1 = __builtin_nontemporal_load(p1 + 1);
            v4f c2 = __builtin_nontemporal_load(p1 + 2);
            v4f c3 = __builtin_nontemporal_load(p1 + 3);

            ssq0 += a0[0]*a0[0] + a0[1]*a0[1] + a0[2]*a0[2] + a0[3]*a0[3]
                  + a1[0]*a1[0] + a1[1]*a1[1] + a1[2]*a1[2] + a1[3]*a1[3]
                  + a2[0]*a2[0] + a2[1]*a2[1] + a2[2]*a2[2] + a2[3]*a2[3]
                  + a3[0]*a3[0] + a3[1]*a3[1] + a3[2]*a3[2] + a3[3]*a3[3];
            ssq1 += c0[0]*c0[0] + c0[1]*c0[1] + c0[2]*c0[2] + c0[3]*c0[3]
                  + c1[0]*c1[0] + c1[1]*c1[1] + c1[2]*c1[2] + c1[3]*c1[3]
                  + c2[0]*c2[0] + c2[1]*c2[1] + c2[2]*c2[2] + c2[3]*c2[3]
                  + c3[0]*c3[0] + c3[1]*c3[1] + c3[2]*c3[2] + c3[3]*c3[3];

            v16bf bm0 = cvt16(a0, a1, a2, a3);
            v16bf bm1 = cvt16(c0, c1, c2, c3);

            const int koff = k0 >> 3;             // immediate-foldable chunk offset
#pragma unroll
            for (int m = 0; m < 4; ++m) {
                AFrag a;
                a.u[0] = qlds[abase[m] + koff];
                a.u[1] = qlds[abase[m] + koff + 1];
                acc[0][m] = __builtin_amdgcn_wmma_f32_16x16x32_bf16(
                    false, a.v, false, bm0, (short)0, acc[0][m], false, false);
                acc[1][m] = __builtin_amdgcn_wmma_f32_16x16x32_bf16(
                    false, a.v, false, bm1, (short)0, acc[1][m], false, false);
            }
        }

        // Row L2 norms: lane pair (L, L^16) covers all K of a row.
        float tot0 = ssq0 + __shfl_xor(ssq0, 16, 32);
        float tot1 = ssq1 + __shfl_xor(ssq1, 16, 32);
        float inv0 = 1.0f / fmaxf(sqrtf(tot0), 1e-8f);
        float inv1 = 1.0f / fmaxf(sqrtf(tot1), 1e-8f);
        const int idx0 = blk * 32 + nrow;
        const int idx1 = idx0 + 16;
#pragma unroll
        for (int m = 0; m < 4; ++m)
#pragma unroll
            for (int r = 0; r < 8; ++r) {
                float s0 = acc[0][m][r] * inv0;
                float s1 = acc[1][m][r] * inv1;
                if (s0 > bestv[m][r]) { bestv[m][r] = s0; besti[m][r] = idx0; }
                if (s1 > bestv[m][r]) { bestv[m][r] = s1; besti[m][r] = idx1; }
            }
    }

    // Butterfly argmax across the 16 lanes of each half, then one atomic/query.
#pragma unroll
    for (int m = 0; m < 4; ++m)
#pragma unroll
        for (int r = 0; r < 8; ++r) {
            float v = bestv[m][r];
            int   i = besti[m][r];
#pragma unroll
            for (int off = 1; off < 16; off <<= 1) {
                float ov = __shfl_xor(v, off, 32);
                int   oi = __shfl_xor(i, off, 32);
                if (ov > v || (ov == v && oi < i)) { v = ov; i = oi; }
            }
            if (nrow == 0) {
                const int q = (m << 4) | (half << 3) | r;   // global query id
                unsigned ub  = __builtin_bit_cast(unsigned, v);
                unsigned key = (ub & 0x80000000u) ? ~ub : (ub | 0x80000000u);
                unsigned long long packed =
                    ((unsigned long long)key << 32) | (unsigned)(~i); // low idx wins
                atomicMax(best + q, packed);
            }
        }
}

// ---------------------------------------------------------------------------
// Kernel 3: stream zeros over the 128 MB one-hot output.
// ---------------------------------------------------------------------------
__global__ void zero_kernel(v4f* __restrict__ out, int n4) {
    int i = blockIdx.x * 256 + threadIdx.x;
    if (i < n4) {
        v4f z = {0.f, 0.f, 0.f, 0.f};
        __builtin_nontemporal_store(z, out + i);
    }
}

// ---------------------------------------------------------------------------
// Kernel 4: scatter the 6400 one-hot 1.0s.
// ---------------------------------------------------------------------------
__global__ void scatter_kernel(const unsigned long long* __restrict__ best,
                               const int* __restrict__ rdb,
                               float* __restrict__ out) {
    int t = blockIdx.x * 256 + threadIdx.x;
    if (t >= QB * LTGT) return;
    int b = t / LTGT, j = t % LTGT;
    unsigned idx = ~(unsigned)(best[b] & 0xFFFFFFFFull);
    int token = rdb[(size_t)idx * LDB + j];
    out[((size_t)b * LTGT + j) * VOCAB + token] = 1.0f;
}

// ---------------------------------------------------------------------------
extern "C" void kernel_launch(void* const* d_in, const int* in_sizes, int n_in,
                              void* d_out, int out_size, void* d_ws, size_t ws_size,
                              hipStream_t stream) {
    const float* feats = (const float*)d_in[0];
    const float* db    = (const float*)d_in[1];
    const int*   rdb   = (const int*)d_in[2];
    float*       out   = (float*)d_out;

    unsigned short*     qbf  = (unsigned short*)d_ws;                        // 64 KB
    unsigned long long* best = (unsigned long long*)((char*)d_ws + QB*QD*2); // 512 B

    const int N    = in_sizes[1] / QD;   // 200000
    const int nblk = N / 32;             // 6250 blocks of 32 DB rows

    norm_kernel<<<QB, 256, 0, stream>>>(feats, qbf, best);

    const int n4 = out_size / 4;
    zero_kernel<<<(n4 + 255) / 256, 256, 0, stream>>>((v4f*)out, n4);

    sim_kernel<<<512, 256, 0, stream>>>(db, (const uint4*)qbf, best, nblk);

    scatter_kernel<<<(QB * LTGT + 255) / 256, 256, 0, stream>>>(best, rdb, out);
}
